// EEGSingleStreamBlock_10754598109575
// MI455X (gfx1250) — matrix operand order repaired and checked by reading, not compiled, not run on hardware
//
#include <hip/hip_runtime.h>
#include <hip/hip_bf16.h>

#define BB    8
#define SS    1024
#define HD    256
#define NH    8
#define DH    32
#define MLP   512
#define C_IN  64
#define COND  32
#define NO    (3*HD + MLP)   // 1280
#define CAT   (HD + MLP)     // 768

typedef __attribute__((ext_vector_type(16))) _Float16 v16h;
typedef __attribute__((ext_vector_type(8)))  _Float16 v8h;
typedef __attribute__((ext_vector_type(8)))  float    v8f;

__device__ __forceinline__ v8f vzero8() {
    v8f z;
#pragma unroll
    for (int i = 0; i < 8; ++i) z[i] = 0.0f;
    return z;
}

__device__ __forceinline__ v8f wmma16(v16h a, v16h b, v8f c) {
    return __builtin_amdgcn_wmma_f32_16x16x32_f16(false, a, false, b, (short)0, c, false, false);
}

// A-operand tile (16 rows x 32 K, f16, row-major with leading dim lda).
// Lane l: row = l%16; element e -> K = (l/16)*16 + e  (16 contiguous halfs)
__device__ __forceinline__ v16h load_a_tile(const _Float16* A, int lda, int lane) {
    int half = lane >> 4, lr = lane & 15;
    const _Float16* p = A + (size_t)lr * lda + half * 16;
    return *(const v16h*)p;
}

// B-operand tile (32 K x 16 N) read from W stored [n][k] row-major, leading dim ldw.
// Lane l: N = l%16; element e -> K = (l/16)*16 + e  (16 contiguous halfs)
__device__ __forceinline__ v16h load_b_tile(const _Float16* W, int ldw, int lane) {
    int half = lane >> 4, lr = lane & 15;
    return *(const v16h*)(W + (size_t)lr * ldw + half * 16);
}

__device__ __forceinline__ v8f wmma_gemm_acc(const _Float16* A, int lda,
                                             const _Float16* W, int ldw,
                                             int K, int lane, v8f acc) {
    for (int k0 = 0; k0 < K; k0 += 32) {
        v16h a = load_a_tile(A + k0, lda, lane);
        v16h b = load_b_tile(W + k0, ldw, lane);
        acc = wmma16(a, b, acc);
    }
    return acc;
}

// ---------------------------------------------------------------- cond path
__global__ void k_cond(const float* cond_vec, const float* ce_w1, const float* ce_b1,
                       const float* ce_w2, const float* ce_b2,
                       const float* mod_w, const float* mod_b,
                       float* shift, float* scale1, float* gate_act) {
    int b = blockIdx.x;
    int c = threadIdx.x;
    __shared__ float t1[HD];
    __shared__ float sf[HD];
    float acc = ce_b1[c];
    for (int k = 0; k < COND; ++k) acc += cond_vec[b * COND + k] * ce_w1[c * COND + k];
    t1[c] = acc / (1.0f + expf(-acc));           // silu
    __syncthreads();
    float cf = ce_b2[c];
    for (int k = 0; k < HD; ++k) cf += t1[k] * ce_w2[c * HD + k];
    sf[c] = cf / (1.0f + expf(-cf));             // silu(cond_feat)
    __syncthreads();
    float m0 = mod_b[c], m1 = mod_b[HD + c], m2 = mod_b[2 * HD + c];
    for (int k = 0; k < HD; ++k) {
        float t = sf[k];
        m0 += t * mod_w[c * HD + k];
        m1 += t * mod_w[(HD + c) * HD + k];
        m2 += t * mod_w[(2 * HD + c) * HD + k];
    }
    shift[b * HD + c]    = m0;
    scale1[b * HD + c]   = 1.0f + m1;
    gate_act[b * HD + c] = 1.0f / (1.0f + expf(-m2));
}

// ---------------------------------------------------------------- f32 -> f16 weight conversion
__global__ void k_cvt(const float* qkvw, const float* opw, const float* finw,
                      _Float16* qkvw16, _Float16* opw16, _Float16* finw16) {
    const int n1 = NO * HD, n2 = HD * CAT, n3 = C_IN * HD;
    int total = n1 + n2 + n3;
    for (int i = blockIdx.x * blockDim.x + threadIdx.x; i < total; i += gridDim.x * blockDim.x) {
        if (i < n1)            qkvw16[i]            = (_Float16)qkvw[i];
        else if (i < n1 + n2)  opw16[i - n1]        = (_Float16)opw[i - n1];
        else                   finw16[i - n1 - n2]  = (_Float16)finw[i - n1 - n2];
    }
}

// ---------------------------------------------------------------- x_h = cp conv, LN, modulation
__global__ __launch_bounds__(256) void k_xh(const float* x, const float* cpw, const float* cpb,
                                            const float* lng, const float* lnb,
                                            const float* scale1, const float* shift,
                                            float* x_h, _Float16* xmod16) {
    int b  = blockIdx.x >> 6;
    int s0 = (blockIdx.x & 63) * 16;
    __shared__ float xs[C_IN][16];
    __shared__ float hls[16][HD];
    __shared__ float mean[16], rinv[16];
    int tid = threadIdx.x;
    for (int i = tid; i < C_IN * 16; i += 256) {
        int kc = i >> 4, sp = i & 15;
        xs[kc][sp] = x[(size_t)b * C_IN * SS + (size_t)kc * SS + s0 + sp];
    }
    __syncthreads();
    int c = tid;
    float h[16];
    float bb = cpb[c];
#pragma unroll
    for (int sp = 0; sp < 16; ++sp) h[sp] = bb;
    for (int kc = 0; kc < C_IN; ++kc) {
        float w = cpw[c * C_IN + kc];
#pragma unroll
        for (int sp = 0; sp < 16; ++sp) h[sp] += w * xs[kc][sp];
    }
#pragma unroll
    for (int sp = 0; sp < 16; ++sp) {
        hls[sp][c] = h[sp];
        x_h[(size_t)b * HD * SS + (size_t)c * SS + s0 + sp] = h[sp];
    }
    __syncthreads();
    if (tid < 16) {
        float s = 0.0f;
        for (int k = 0; k < HD; ++k) s += hls[tid][k];
        float m = s / HD;
        float v = 0.0f;
        for (int k = 0; k < HD; ++k) { float d = hls[tid][k] - m; v += d * d; }
        mean[tid] = m;
        rinv[tid] = rsqrtf(v / HD + 1e-5f);
    }
    __syncthreads();
    float g = lng[c], be = lnb[c], sc = scale1[b * HD + c], sh = shift[b * HD + c];
#pragma unroll
    for (int sp = 0; sp < 16; ++sp) {
        float xn = (h[sp] - mean[sp]) * rinv[sp] * g + be;
        xmod16[((size_t)(b * SS + s0 + sp)) * HD + c] = (_Float16)(xn * sc + sh);
    }
}

// ---------------------------------------------------------------- qkv GEMM (WMMA): 8192 x 1280 x 256
__global__ __launch_bounds__(256) void k_qkv(const _Float16* A, const _Float16* W,
                                             const float* bias, _Float16* out) {
    int wave = (blockIdx.x * blockDim.x + threadIdx.x) >> 5;
    int lane = threadIdx.x & 31;
    const int TN = NO / 16;                  // 80
    int tm = wave / TN, tn = wave % TN;
    v8f acc = vzero8();
    acc = wmma_gemm_acc(A + (size_t)tm * 16 * HD, HD, W + (size_t)tn * 16 * HD, HD, HD, lane, acc);
    int half = lane >> 4, lr = lane & 15;
    int n = tn * 16 + lr;
    float bv = bias[n];
#pragma unroll
    for (int r = 0; r < 8; ++r) {
        int m = tm * 16 + r + half * 8;
        out[(size_t)m * NO + n] = (_Float16)(acc[r] + bv);
    }
}

// ---------------------------------------------------------------- xp = pi projection of v (3 ch)
__global__ void k_xp(const _Float16* qkv16, const float* piw, const float* pib, float* xp) {
    int i = blockIdx.x * 256 + threadIdx.x;      // 8192 = B*S
    int b = i >> 10, s = i & 1023;
    float a0 = pib[0], a1 = pib[1], a2 = pib[2];
    const _Float16* vp = qkv16 + (size_t)i * NO + 2 * HD;
    for (int c = 0; c < HD; ++c) {
        float vv = (float)vp[c];
        a0 += piw[c] * vv;
        a1 += piw[HD + c] * vv;
        a2 += piw[2 * HD + c] * vv;
    }
    xp[(b * 3 + 0) * SS + s] = a0;
    xp[(b * 3 + 1) * SS + s] = a1;
    xp[(b * 3 + 2) * SS + s] = a2;
}

// ---------------------------------------------------------------- mt layers: conv1d -> LN(len) -> elu -> mean
__global__ void k_mt(const float* xp,
                     const float* c2w, const float* c2b,
                     const float* c4w, const float* c4b,
                     const float* c6w, const float* c6b,
                     float* attw) {
    int b = blockIdx.x / 3, j = blockIdx.x % 3;
    const float* w; const float* bi; int K, pad;
    if (j == 0)      { w = c2w; bi = c2b; K = 2; pad = 0; }
    else if (j == 1) { w = c4w; bi = c4b; K = 4; pad = 1; }
    else             { w = c6w; bi = c6b; K = 6; pad = 2; }
    const int L = 1023;
    __shared__ float ybuf[1023];
    __shared__ float red[256];
    const float* xr = xp + (b * 3 + j) * SS;
    int tid = threadIdx.x;
    float lsum = 0.0f;
    for (int l = tid; l < L; l += 256) {
        float acc = bi[0];
        for (int t = 0; t < K; ++t) {
            int sidx = l - pad + t;
            float xv = (sidx >= 0 && sidx < SS) ? xr[sidx] : 0.0f;
            acc += w[t] * xv;
        }
        ybuf[l] = acc;
        lsum += acc;
    }
    red[tid] = lsum; __syncthreads();
    for (int st = 128; st > 0; st >>= 1) { if (tid < st) red[tid] += red[tid + st]; __syncthreads(); }
    float mval = red[0] / L;
    __syncthreads();
    float lv = 0.0f;
    for (int l = tid; l < L; l += 256) { float d = ybuf[l] - mval; lv += d * d; }
    red[tid] = lv; __syncthreads();
    for (int st = 128; st > 0; st >>= 1) { if (tid < st) red[tid] += red[tid + st]; __syncthreads(); }
    float rv = rsqrtf(red[0] / L + 1e-5f);
    __syncthreads();
    float ls2 = 0.0f;
    for (int l = tid; l < L; l += 256) {
        float yn = (ybuf[l] - mval) * rv;
        ls2 += (yn > 0.0f) ? yn : (expf(yn) - 1.0f);     // elu
    }
    red[tid] = ls2; __syncthreads();
    for (int st = 128; st > 0; st >>= 1) { if (tid < st) red[tid] += red[tid + st]; __syncthreads(); }
    if (tid == 0) attw[b * 3 + j] = red[0] / L;
}

// ---------------------------------------------------------------- v_new = po(xp*attw) + v, stored [b][c][s]
__global__ void k_vnew(const _Float16* qkv16, const float* xp, const float* attw,
                       const float* po_w, const float* po_b, _Float16* vnew16) {
    int i = blockIdx.x * 256 + threadIdx.x;      // 8*256*1024
    int s = i & 1023;
    int c = (i >> 10) & 255;
    int b = i >> 18;
    float vv = (float)qkv16[((size_t)(b * SS + s)) * NO + 2 * HD + c];
    float acc = po_b[c] + vv;
#pragma unroll
    for (int j = 0; j < 3; ++j)
        acc += po_w[c * 3 + j] * attw[b * 3 + j] * xp[(b * 3 + j) * SS + s];
    vnew16[((size_t)(b * HD + c)) * SS + s] = (_Float16)acc;
}

// ---------------------------------------------------------------- per-head LN of q and k
__global__ void k_qkln(const _Float16* qkv16, const float* qng, const float* qnb,
                       const float* kng, const float* knb, _Float16* qln, _Float16* kln) {
    int i = blockIdx.x * 256 + threadIdx.x;      // (b*NH+h)*S + s  = 65536
    int s = i & 1023; int bh = i >> 10; int h = bh & 7; int b = bh >> 3;
    const _Float16* qp = qkv16 + ((size_t)(b * SS + s)) * NO + h * DH;
    float vals[DH];
    // q
    float sum = 0.0f;
    for (int d = 0; d < DH; ++d) { vals[d] = (float)qp[d]; sum += vals[d]; }
    float m = sum / DH, var = 0.0f;
    for (int d = 0; d < DH; ++d) { float dd = vals[d] - m; var += dd * dd; }
    float ri = rsqrtf(var / DH + 1e-5f);
    _Float16* qo = qln + (size_t)i * DH;
    for (int d = 0; d < DH; ++d) qo[d] = (_Float16)((vals[d] - m) * ri * qng[d] + qnb[d]);
    // k (channels HD + h*DH + d)
    const _Float16* kp = qp + HD;
    sum = 0.0f;
    for (int d = 0; d < DH; ++d) { vals[d] = (float)kp[d]; sum += vals[d]; }
    m = sum / DH; var = 0.0f;
    for (int d = 0; d < DH; ++d) { float dd = vals[d] - m; var += dd * dd; }
    ri = rsqrtf(var / DH + 1e-5f);
    _Float16* ko = kln + (size_t)i * DH;
    for (int d = 0; d < DH; ++d) ko[d] = (_Float16)((vals[d] - m) * ri * kng[d] + knb[d]);
}

// ---------------------------------------------------------------- exact GELU of mlp_in into cat buffer
__global__ void k_gelu(const _Float16* qkv16, _Float16* cat16) {
    int i = blockIdx.x * 256 + threadIdx.x;      // 8192*512
    int bs = i >> 9, j = i & 511;
    float xv = (float)qkv16[(size_t)bs * NO + 3 * HD + j];
    float g = 0.5f * xv * (1.0f + erff(xv * 0.70710678118654752f));
    cat16[(size_t)bs * CAT + HD + j] = (_Float16)g;
}

// ---------------------------------------------------------------- flash attention, one wave / 16-query tile
__global__ __launch_bounds__(32) void k_attn(const _Float16* qln, const _Float16* kln,
                                             const _Float16* vnew, _Float16* cat16) {
    __shared__ __align__(32) _Float16 pbuf[16 * 32];
    int blk = blockIdx.x;                        // b*NH*64 + h*64 + ts
    int ts = blk & 63; int bh = blk >> 6; int h = bh & 7; int b = bh >> 3;
    int lane = threadIdx.x & 31, half = lane >> 4, lr = lane & 15;
    const _Float16* qbase = qln + (size_t)bh * SS * DH + (size_t)ts * 16 * DH;
    const _Float16* kbase = kln + (size_t)bh * SS * DH;
    const _Float16* vbase = vnew + (size_t)(b * HD + h * DH) * SS;

    v16h aq = load_a_tile(qbase, DH, lane);      // loop-invariant Q tile
    float rm[8], rl[8];
    v8f o0 = vzero8(), o1 = vzero8();
#pragma unroll
    for (int r = 0; r < 8; ++r) { rm[r] = -1e30f; rl[r] = 0.0f; }
    const float scl = 0.17677669529663687f;      // 1/sqrt(32)

    for (int t0 = 0; t0 < SS; t0 += 32) {
        v16h bk0 = load_b_tile(kbase + (size_t)t0 * DH, DH, lane);
        v16h bk1 = load_b_tile(kbase + (size_t)(t0 + 16) * DH, DH, lane);
        v8f s0 = wmma16(aq, bk0, vzero8());
        v8f s1 = wmma16(aq, bk1, vzero8());
        float p0[8], p1[8];
#pragma unroll
        for (int r = 0; r < 8; ++r) {
            float a = s0[r] * scl, c = s1[r] * scl;
            float mx = fmaxf(a, c);
            mx = fmaxf(mx, __shfl_xor(mx, 1));
            mx = fmaxf(mx, __shfl_xor(mx, 2));
            mx = fmaxf(mx, __shfl_xor(mx, 4));
            mx = fmaxf(mx, __shfl_xor(mx, 8));
            float mnew = fmaxf(rm[r], mx);
            float corr = expf(rm[r] - mnew);
            rm[r] = mnew;
            float e0 = expf(a - mnew), e1 = expf(c - mnew);
            p0[r] = e0; p1[r] = e1;
            float tsum = e0 + e1;
            tsum += __shfl_xor(tsum, 1);
            tsum += __shfl_xor(tsum, 2);
            tsum += __shfl_xor(tsum, 4);
            tsum += __shfl_xor(tsum, 8);
            rl[r] = rl[r] * corr + tsum;
            o0[r] *= corr; o1[r] *= corr;
        }
        // C-layout -> A-layout via LDS: pbuf[m][t_local], m = r + 8*half
#pragma unroll
        for (int r = 0; r < 8; ++r) {
            int m = r + 8 * half;
            pbuf[m * 32 + lr]      = (_Float16)p0[r];
            pbuf[m * 32 + 16 + lr] = (_Float16)p1[r];
        }
        __syncthreads();
        v16h ap  = load_a_tile(pbuf, 32, lane);
        v16h bv0 = load_b_tile(vbase + t0, SS, lane);               // d = 0..15
        v16h bv1 = load_b_tile(vbase + (size_t)16 * SS + t0, SS, lane); // d = 16..31
        o0 = wmma16(ap, bv0, o0);
        o1 = wmma16(ap, bv1, o1);
        __syncthreads();
    }
    int s0q = ts * 16;
#pragma unroll
    for (int r = 0; r < 8; ++r) {
        int m = r + 8 * half;
        float inv = 1.0f / rl[r];
        size_t base = ((size_t)(b * SS + s0q + m)) * CAT + h * DH;
        cat16[base + lr]      = (_Float16)(o0[r] * inv);
        cat16[base + 16 + lr] = (_Float16)(o1[r] * inv);
    }
}

// ---------------------------------------------------------------- output proj GEMM + residual/gate
__global__ __launch_bounds__(256) void k_out(const _Float16* A, const _Float16* W,
                                             const float* op_b, const float* x_h,
                                             const float* gate_act, _Float16* xafter16) {
    int wave = (blockIdx.x * blockDim.x + threadIdx.x) >> 5;
    int lane = threadIdx.x & 31;
    const int TN = HD / 16;                      // 16
    int tm = wave / TN, tn = wave % TN;
    v8f acc = vzero8();
    acc = wmma_gemm_acc(A + (size_t)tm * 16 * CAT, CAT, W + (size_t)tn * 16 * CAT, CAT, CAT, lane, acc);
    int half = lane >> 4, lr = lane & 15;
    int n = tn * 16 + lr;
    float ob = op_b[n];
#pragma unroll
    for (int r = 0; r < 8; ++r) {
        int m = tm * 16 + r + half * 8;
        int b = m >> 10, s = m & 1023;
        float xh = x_h[(size_t)b * HD * SS + (size_t)n * SS + s];
        float ga = gate_act[b * HD + n];
        xafter16[(size_t)m * HD + n] = (_Float16)(xh + ga * (acc[r] + ob));
    }
}

// ---------------------------------------------------------------- final proj GEMM -> d_out (f32)
__global__ __launch_bounds__(256) void k_fin(const _Float16* A, const _Float16* W,
                                             const float* fin_b, float* out) {
    int wave = (blockIdx.x * blockDim.x + threadIdx.x) >> 5;
    int lane = threadIdx.x & 31;
    const int TN = C_IN / 16;                    // 4
    int tm = wave / TN, tn = wave % TN;
    v8f acc = vzero8();
    acc = wmma_gemm_acc(A + (size_t)tm * 16 * HD, HD, W + (size_t)tn * 16 * HD, HD, HD, lane, acc);
    int half = lane >> 4, lr = lane & 15;
    int n = tn * 16 + lr;
    float bv = fin_b[n];
#pragma unroll
    for (int r = 0; r < 8; ++r) {
        int m = tm * 16 + r + half * 8;
        int b = m >> 10, s = m & 1023;
        out[(size_t)b * C_IN * SS + (size_t)n * SS + s] = acc[r] + bv;
    }
}

extern "C" void kernel_launch(void* const* d_in, const int* in_sizes, int n_in,
                              void* d_out, int out_size, void* d_ws, size_t ws_size,
                              hipStream_t stream) {
    const float* x        = (const float*)d_in[0];
    const float* cond_vec = (const float*)d_in[1];
    const float* ce_w1    = (const float*)d_in[2];
    const float* ce_b1    = (const float*)d_in[3];
    const float* ce_w2    = (const float*)d_in[4];
    const float* ce_b2    = (const float*)d_in[5];
    const float* mod_w    = (const float*)d_in[6];
    const float* mod_b    = (const float*)d_in[7];
    const float* cp_w     = (const float*)d_in[8];
    const float* cp_b     = (const float*)d_in[9];
    const float* ln_g     = (const float*)d_in[10];
    const float* ln_b     = (const float*)d_in[11];
    const float* qkv_w    = (const float*)d_in[12];
    const float* qkv_b    = (const float*)d_in[13];
    const float* pi_w     = (const float*)d_in[14];
    const float* pi_b     = (const float*)d_in[15];
    const float* po_w     = (const float*)d_in[16];
    const float* po_b     = (const float*)d_in[17];
    const float* c2_w     = (const float*)d_in[18];
    const float* c2_b     = (const float*)d_in[19];
    const float* c4_w     = (const float*)d_in[20];
    const float* c4_b     = (const float*)d_in[21];
    const float* c6_w     = (const float*)d_in[22];
    const float* c6_b     = (const float*)d_in[23];
    const float* qn_g     = (const float*)d_in[24];
    const float* qn_b     = (const float*)d_in[25];
    const float* kn_g     = (const float*)d_in[26];
    const float* kn_b     = (const float*)d_in[27];
    const float* op_w     = (const float*)d_in[28];
    const float* op_b     = (const float*)d_in[29];
    const float* fin_w    = (const float*)d_in[30];
    const float* fin_b    = (const float*)d_in[31];
    float* out = (float*)d_out;

    char* ws = (char*)d_ws;
    size_t off = 0;
    auto take = [&](size_t bytes) -> void* {
        void* p = ws + off;
        off += (bytes + 255) & ~(size_t)255;
        return p;
    };
    float*    shift    = (float*)take((size_t)BB * HD * 4);
    float*    scale1   = (float*)take((size_t)BB * HD * 4);
    float*    gate_act = (float*)take((size_t)BB * HD * 4);
    float*    attw     = (float*)take((size_t)BB * 3 * 4);
    _Float16* qkvw16   = (_Float16*)take((size_t)NO * HD * 2);
    _Float16* opw16    = (_Float16*)take((size_t)HD * CAT * 2);
    _Float16* finw16   = (_Float16*)take((size_t)C_IN * HD * 2);
    float*    x_h      = (float*)take((size_t)BB * HD * SS * 4);
    _Float16* xmod16   = (_Float16*)take((size_t)BB * SS * HD * 2);
    _Float16* qkv16    = (_Float16*)take((size_t)BB * SS * NO * 2);
    float*    xp       = (float*)take((size_t)BB * 3 * SS * 4);
    _Float16* vnew16   = (_Float16*)take((size_t)BB * HD * SS * 2);
    _Float16* qln16    = (_Float16*)take((size_t)BB * NH * SS * DH * 2);
    _Float16* kln16    = (_Float16*)take((size_t)BB * NH * SS * DH * 2);
    _Float16* cat16    = (_Float16*)take((size_t)BB * SS * CAT * 2);
    _Float16* xafter16 = (_Float16*)take((size_t)BB * SS * HD * 2);

    k_cond<<<BB, HD, 0, stream>>>(cond_vec, ce_w1, ce_b1, ce_w2, ce_b2, mod_w, mod_b,
                                  shift, scale1, gate_act);
    k_cvt<<<1024, 256, 0, stream>>>(qkv_w, op_w, fin_w, qkvw16, opw16, finw16);
    k_xh<<<BB * (SS / 16), 256, 0, stream>>>(x, cp_w, cp_b, ln_g, ln_b, scale1, shift,
                                             x_h, xmod16);
    // 512*80 = 40960 wave-tiles, 8 waves/block
    k_qkv<<<5120, 256, 0, stream>>>(xmod16, qkvw16, qkv_b, qkv16);
    k_xp<<<32, 256, 0, stream>>>(qkv16, pi_w, pi_b, xp);
    k_mt<<<BB * 3, 256, 0, stream>>>(xp, c2_w, c2_b, c4_w, c4_b, c6_w, c6_b, attw);
    k_vnew<<<8192, 256, 0, stream>>>(qkv16, xp, attw, po_w, po_b, vnew16);
    k_qkln<<<256, 256, 0, stream>>>(qkv16, qn_g, qn_b, kn_g, kn_b, qln16, kln16);
    k_gelu<<<16384, 256, 0, stream>>>(qkv16, cat16);
    k_attn<<<BB * NH * (SS / 16), 32, 0, stream>>>(qln16, kln16, vnew16, cat16);
    // 512*16 = 8192 tiles, 8 waves/block
    k_out<<<1024, 256, 0, stream>>>(cat16, opw16, op_b, x_h, gate_act, xafter16);
    // 512*4 = 2048 tiles
    k_fin<<<256, 256, 0, stream>>>(xafter16, finw16, fin_b, out);
}
